// VQVitModelPlus_85873576116616
// MI455X (gfx1250) — matrix operand order repaired and checked by reading, not compiled, not run on hardware
//
#include <hip/hip_runtime.h>
#include <hip/hip_bf16.h>
#include <math.h>

// ---------------- problem constants ----------------
#define N_E    16384
#define E_DIM  32
#define HW     256            // 16*16
#define T_TOK  8192           // 32 * 256 tokens
#define OUT_N  (T_TOK * E_DIM)

// ---------------- tiling ----------------
#define CHUNK   512                   // codebook rows per LDS buffer (32 KB)
#define NCHUNK  (N_E / CHUNK)         // 32
#define TILES   (CHUNK / 16)          // 32 WMMA tiles per chunk
#define WAVES   8                     // waves per block (256 threads)
#define BLOCKS  (T_TOK / 16 / WAVES)  // 64 blocks, one 16-token strip per wave
#define LPT     8                     // async b128 loads per thread per chunk

typedef __attribute__((ext_vector_type(16))) _Float16 v16h;
typedef __attribute__((ext_vector_type(8)))  _Float16 v8h;
typedef __attribute__((ext_vector_type(8)))  float    v8f;
typedef int v4i __attribute__((__vector_size__(16)));   // matches builtin prototype

#if defined(__HIP_DEVICE_COMPILE__)
#if __has_builtin(__builtin_amdgcn_global_load_async_to_lds_b128) && \
    __has_builtin(__builtin_amdgcn_s_wait_asynccnt)
#define USE_ASYNC_LDS 1
#endif
#endif

// ---------------- issue one chunk of codebook staging (global -> LDS) ----------------
// Async path: fire-and-forget (caller waits with s_wait_asynccnt).
// Fallback: synchronous copy (caller's __syncthreads() publishes it).
__device__ __forceinline__ void stage_issue(const _Float16* __restrict__ src,
                                            _Float16* __restrict__ dst, int tid) {
#ifdef USE_ASYNC_LDS
#pragma unroll
  for (int k = 0; k < LPT; ++k) {
    int i = tid + k * 256;
    __builtin_amdgcn_global_load_async_to_lds_b128(
        (__attribute__((address_space(1))) v4i*)(src + (size_t)i * 8),
        (__attribute__((address_space(3))) v4i*)(dst + (size_t)i * 8),
        0, 0);
  }
#else
  const float4* s4 = (const float4*)src;
  float4* d4 = (float4*)dst;
#pragma unroll
  for (int k = 0; k < LPT; ++k) { int i = tid + k * 256; d4[i] = s4[i]; }
#endif
}

__device__ __forceinline__ void stage_wait_current() {
#ifdef USE_ASYNC_LDS
  __builtin_amdgcn_s_wait_asynccnt(LPT);   // next chunk's LPT may stay in flight
#endif
}
__device__ __forceinline__ void stage_wait_all() {
#ifdef USE_ASYNC_LDS
  __builtin_amdgcn_s_wait_asynccnt(0);
#endif
}

// ---------------- B fragment from LDS: 32x16 f16, lane = code col ----------------
__device__ __forceinline__ v16h load_bfrag(const _Float16* lds_emb, int ti,
                                           int col, int halfId) {
  const _Float16* bp = lds_emb + (size_t)(ti * 16 + col) * E_DIM + halfId * 16;
  v8h b0 = *(const v8h*)bp;
  v8h b1 = *(const v8h*)(bp + 8);
  return __builtin_shufflevector(b0, b1, 0,1,2,3,4,5,6,7,8,9,10,11,12,13,14,15);
}

// ---------------- prep: normalize z tokens ----------------
__global__ void vq_prep_z(const float* __restrict__ z,
                          float* __restrict__ zbn,
                          _Float16* __restrict__ zf16) {
  int t = blockIdx.x * blockDim.x + threadIdx.x;
  if (t >= T_TOK) return;
  int b = t >> 8, hw = t & 255;
  const float* base = z + (size_t)b * (E_DIM * HW) + hw;   // (b,c,h,w), stride HW over c
  float v[E_DIM];
  float ss = 0.f;
#pragma unroll
  for (int c = 0; c < E_DIM; ++c) { float x = base[(size_t)c * HW]; v[c] = x; ss += x * x; }
  float inv = 1.f / fmaxf(sqrtf(ss), 1e-12f);
#pragma unroll
  for (int c = 0; c < E_DIM; ++c) {
    float nv = v[c] * inv;
    zbn[(size_t)t * E_DIM + c]  = nv;
    zf16[(size_t)t * E_DIM + c] = (_Float16)nv;
  }
}

// ---------------- prep: normalize codebook ----------------
__global__ void vq_prep_emb(const float* __restrict__ emb,
                            float* __restrict__ embn,
                            _Float16* __restrict__ embf16) {
  int n = blockIdx.x * blockDim.x + threadIdx.x;
  if (n >= N_E) return;
  const float* r = emb + (size_t)n * E_DIM;
  float v[E_DIM];
  float ss = 0.f;
#pragma unroll
  for (int c = 0; c < E_DIM; ++c) { float x = r[c]; v[c] = x; ss += x * x; }
  float inv = 1.f / fmaxf(sqrtf(ss), 1e-12f);
#pragma unroll
  for (int c = 0; c < E_DIM; ++c) {
    float nv = v[c] * inv;
    embn[(size_t)n * E_DIM + c]   = nv;
    embf16[(size_t)n * E_DIM + c] = (_Float16)nv;
  }
}

// ---------------- zero accumulators ----------------
__global__ void vq_zero(float* __restrict__ avg_probs, float* __restrict__ acc) {
  int i = blockIdx.x * blockDim.x + threadIdx.x;
  if (i < N_E) avg_probs[i] = 0.f;
  if (i < 2)   acc[i] = 0.f;
}

// ---------------- main: fused GEMM + argmax + softmax stats + avg_probs ----------------
__global__ __launch_bounds__(256)
void vq_main(const _Float16* __restrict__ zf16,
             const _Float16* __restrict__ embf16,
             int*   __restrict__ tok_idx,
             float* __restrict__ avg_probs,
             float* __restrict__ acc) {         // acc[0]=vq sum, acc[1]=sample-entropy sum
  __shared__ __align__(16) _Float16 buf0[CHUNK * E_DIM];   // 32 KB double buffer A
  __shared__ __align__(16) _Float16 buf1[CHUNK * E_DIM];   // 32 KB double buffer B
  __shared__ float lds_acc[CHUNK];                         // 2 KB prob accumulator

  const int tid    = threadIdx.x;
  const int wave   = tid >> 5;
  const int lane   = tid & 31;
  const int halfId = lane >> 4;       // wave32 lane half selects K range
  const int col    = lane & 15;       // C/D layout: lane&15 is the N column
  const int strip  = blockIdx.x * WAVES + wave;   // 16-token row strip

  // --- A fragment: 16x32 f16, lane = token row M, K split by lane half ---
  const _Float16* ap = zf16 + ((size_t)(strip * 16 + col)) * E_DIM + halfId * 16;
  v8h alo = *(const v8h*)ap;           // K = h*16 .. h*16+7
  v8h ahi = *(const v8h*)(ap + 8);     // K = h*16+8 .. h*16+15
  v16h afrag = __builtin_shufflevector(alo, ahi,
      0,1,2,3,4,5,6,7,8,9,10,11,12,13,14,15);

  // ================= phase 1: raw-dot max & argmax (no exp needed) =================
  float cmax[8]; int bidx[8];
#pragma unroll
  for (int j = 0; j < 8; ++j) { cmax[j] = -1e30f; bidx[j] = 0; }

  stage_issue(embf16, buf0, tid);
  for (int ch = 0; ch < NCHUNK; ++ch) {
    const int cbase = ch * CHUNK;
    const _Float16* cur = (ch & 1) ? buf1 : buf0;
    _Float16*       nxt = (ch & 1) ? buf0 : buf1;
    if (ch + 1 < NCHUNK) {
      stage_issue(embf16 + (size_t)(cbase + CHUNK) * E_DIM, nxt, tid);
      stage_wait_current();
    } else {
      stage_wait_all();
    }
    if (ch + 2 < NCHUNK)
      __builtin_prefetch(embf16 + (size_t)(cbase + 2 * CHUNK) * E_DIM, 0, 1);
    __syncthreads();

    v16h bf0 = load_bfrag(cur, 0, col, halfId);
    v16h bf1 = load_bfrag(cur, 1, col, halfId);
    for (int ti = 0; ti < TILES; ti += 2) {
      v8f c0 = {}, c1 = {};
      c0 = __builtin_amdgcn_wmma_f32_16x16x32_f16(false, afrag, false, bf0,
                                                  (short)0, c0, false, false);
      c1 = __builtin_amdgcn_wmma_f32_16x16x32_f16(false, afrag, false, bf1,
                                                  (short)0, c1, false, false);
      const int tn = (ti + 2) & (TILES - 1);     // pipeline next pair's LDS loads
      bf0 = load_bfrag(cur, tn,     col, halfId);
      bf1 = load_bfrag(cur, tn + 1, col, halfId);
      const int n0 = cbase + ti * 16 + col;
#pragma unroll
      for (int j = 0; j < 8; ++j) {
        if (c0[j] > cmax[j]) { cmax[j] = c0[j]; bidx[j] = n0; }
        if (c1[j] > cmax[j]) { cmax[j] = c1[j]; bidx[j] = n0 + 16; }
      }
    }
    __syncthreads();
  }

  // merge max/argmax across the 16 lanes sharing each token row
#pragma unroll
  for (int j = 0; j < 8; ++j) {
    for (int msk = 1; msk < 16; msk <<= 1) {
      float om = __shfl_xor(cmax[j], msk, 32);
      int   oi = __shfl_xor(bidx[j], msk, 32);
      bool take = (om > cmax[j]) || ((om == cmax[j]) && (oi < bidx[j]));
      bidx[j] = take ? oi : bidx[j];
      cmax[j] = fmaxf(cmax[j], om);
    }
  }
  float mx[8];
#pragma unroll
  for (int j = 0; j < 8; ++j) mx[j] = 200.f * cmax[j];   // logit max

  if (col == 0) {   // lanes 0 and 16 own tokens M = halfId*8 + j
#pragma unroll
    for (int j = 0; j < 8; ++j)
      tok_idx[strip * 16 + halfId * 8 + j] = bidx[j];
  }

  // ================= phase 2: sum-exp and sum d*e (one exp / score) =================
  float s1[8], s2[8];
#pragma unroll
  for (int j = 0; j < 8; ++j) { s1[j] = 0.f; s2[j] = 0.f; }

  stage_issue(embf16, buf0, tid);
  for (int ch = 0; ch < NCHUNK; ++ch) {
    const int cbase = ch * CHUNK;
    const _Float16* cur = (ch & 1) ? buf1 : buf0;
    _Float16*       nxt = (ch & 1) ? buf0 : buf1;
    if (ch + 1 < NCHUNK) {
      stage_issue(embf16 + (size_t)(cbase + CHUNK) * E_DIM, nxt, tid);
      stage_wait_current();
    } else {
      stage_wait_all();
    }
    __syncthreads();

    v16h bf0 = load_bfrag(cur, 0, col, halfId);
    v16h bf1 = load_bfrag(cur, 1, col, halfId);
    for (int ti = 0; ti < TILES; ti += 2) {
      v8f c0 = {}, c1 = {};
      c0 = __builtin_amdgcn_wmma_f32_16x16x32_f16(false, afrag, false, bf0,
                                                  (short)0, c0, false, false);
      c1 = __builtin_amdgcn_wmma_f32_16x16x32_f16(false, afrag, false, bf1,
                                                  (short)0, c1, false, false);
      const int tn = (ti + 2) & (TILES - 1);
      bf0 = load_bfrag(cur, tn,     col, halfId);
      bf1 = load_bfrag(cur, tn + 1, col, halfId);
#pragma unroll
      for (int j = 0; j < 8; ++j) {
        float d0 = fmaf(200.f, c0[j], -mx[j]);
        float e0 = __expf(d0);
        s1[j] += e0;
        s2[j] = fmaf(d0, e0, s2[j]);
        float d1 = fmaf(200.f, c1[j], -mx[j]);
        float e1 = __expf(d1);
        s1[j] += e1;
        s2[j] = fmaf(d1, e1, s2[j]);
      }
    }
    __syncthreads();
  }

  // plain-add merge (all lanes used the same per-token max)
#pragma unroll
  for (int j = 0; j < 8; ++j) {
    for (int msk = 1; msk < 16; msk <<= 1) {
      s1[j] += __shfl_xor(s1[j], msk, 32);
      s2[j] += __shfl_xor(s2[j], msk, 32);
    }
  }
  float r1[8];
#pragma unroll
  for (int j = 0; j < 8; ++j) r1[j] = 1.f / s1[j];

  if (col == 0) {
    float loc = 0.f;
#pragma unroll
    for (int j = 0; j < 8; ++j)
      loc += s2[j] * r1[j] - __logf(s1[j]);     // sum_n p*logp per token
    atomicAdd(&acc[1], loc);
  }

  // ================= phase 3: accumulate avg_probs =================
  stage_issue(embf16, buf0, tid);
  for (int ch = 0; ch < NCHUNK; ++ch) {
    const int cbase = ch * CHUNK;
    const _Float16* cur = (ch & 1) ? buf1 : buf0;
    _Float16*       nxt = (ch & 1) ? buf0 : buf1;
    if (ch + 1 < NCHUNK) {
      stage_issue(embf16 + (size_t)(cbase + CHUNK) * E_DIM, nxt, tid);
      stage_wait_current();
    } else {
      stage_wait_all();
    }
    for (int i = tid; i < CHUNK; i += 256) lds_acc[i] = 0.f;
    __syncthreads();

    v16h bf0 = load_bfrag(cur, 0, col, halfId);
    v16h bf1 = load_bfrag(cur, 1, col, halfId);
    for (int ti = 0; ti < TILES; ti += 2) {
      v8f c0 = {}, c1 = {};
      c0 = __builtin_amdgcn_wmma_f32_16x16x32_f16(false, afrag, false, bf0,
                                                  (short)0, c0, false, false);
      c1 = __builtin_amdgcn_wmma_f32_16x16x32_f16(false, afrag, false, bf1,
                                                  (short)0, c1, false, false);
      const int tn = (ti + 2) & (TILES - 1);
      bf0 = load_bfrag(cur, tn,     col, halfId);
      bf1 = load_bfrag(cur, tn + 1, col, halfId);
      // the 8 rows in a lane share one code column: reduce in regs, 1 ds atomic each
      float p0 = 0.f, p1 = 0.f;
#pragma unroll
      for (int j = 0; j < 8; ++j) {
        p0 = fmaf(__expf(fmaf(200.f, c0[j], -mx[j])), r1[j], p0);
        p1 = fmaf(__expf(fmaf(200.f, c1[j], -mx[j])), r1[j], p1);
      }
      atomicAdd(&lds_acc[ti * 16 + col], p0);        // ds_add_f32
      atomicAdd(&lds_acc[(ti + 1) * 16 + col], p1);
    }
    __syncthreads();
    for (int i = tid; i < CHUNK; i += 256) atomicAdd(&avg_probs[cbase + i], lds_acc[i]);
    __syncthreads();
  }
}

// ---------------- gather output + vq loss ----------------
__global__ void vq_gather(const int* __restrict__ tok_idx,
                          const float* __restrict__ embn,
                          const float* __restrict__ zbn,
                          float* __restrict__ out,
                          float* __restrict__ acc) {
  int t = blockIdx.x * blockDim.x + threadIdx.x;
  if (t >= T_TOK) return;
  int b = t >> 8, hw = t & 255;
  int n = tok_idx[t];
  const float* er = embn + (size_t)n * E_DIM;
  const float* zr = zbn  + (size_t)t * E_DIM;
  float sq = 0.f;
#pragma unroll
  for (int c = 0; c < E_DIM; ++c) {
    float e = er[c];
    float d = e - zr[c];
    sq += d * d;
    out[(size_t)b * (E_DIM * HW) + (size_t)c * HW + hw] = e;  // straight-through == z_q
  }
  atomicAdd(&acc[0], sq);
}

// ---------------- finalize scalars ----------------
__global__ void vq_finalize(const float* __restrict__ avg_probs,
                            const float* __restrict__ acc,
                            float* __restrict__ out) {
  __shared__ float red[256];
  int tid = threadIdx.x;
  float part = 0.f;
  for (int i = tid; i < N_E; i += 256) {
    float a = avg_probs[i] * (1.f / (float)T_TOK);
    part -= a * __logf(a + 1e-5f);
  }
  red[tid] = part;
  __syncthreads();
  for (int s = 128; s > 0; s >>= 1) {
    if (tid < s) red[tid] += red[tid + s];
    __syncthreads();
  }
  if (tid == 0) {
    float avg_ent    = red[0];
    float sample_ent = -acc[1] / (float)T_TOK;
    float vq         = acc[0] / (float)(T_TOK * E_DIM);
    out[OUT_N + 0] = vq;
    out[OUT_N + 1] = 0.25f * vq;                       // BETA * same forward MSE
    out[OUT_N + 2] = 0.1f * (sample_ent - avg_ent);    // ENT_RATIO
  }
}

// ---------------- host launcher ----------------
extern "C" void kernel_launch(void* const* d_in, const int* in_sizes, int n_in,
                              void* d_out, int out_size, void* d_ws, size_t ws_size,
                              hipStream_t stream) {
  (void)in_sizes; (void)n_in; (void)out_size; (void)ws_size;
  const float* z   = (const float*)d_in[0];
  const float* emb = (const float*)d_in[1];
  float* out = (float*)d_out;

  char* ws = (char*)d_ws;
  _Float16* zf16   = (_Float16*)(ws + 0);          // 512 KB
  _Float16* embf16 = (_Float16*)(ws + 524288);     // 1 MB
  float*    zbn    = (float*)(ws + 1572864);       // 1 MB
  float*    embn   = (float*)(ws + 2621440);       // 2 MB
  int*      tidx   = (int*)(ws + 4718592);         // 32 KB
  float*    avgp   = (float*)(ws + 4751360);       // 64 KB
  float*    acc    = (float*)(ws + 4816896);       // 2 floats

  vq_zero    <<<N_E / 256, 256, 0, stream>>>(avgp, acc);
  vq_prep_z  <<<T_TOK / 256, 256, 0, stream>>>(z, zbn, zf16);
  vq_prep_emb<<<N_E / 256, 256, 0, stream>>>(emb, embn, embf16);
  vq_main    <<<BLOCKS, 256, 0, stream>>>(zf16, embf16, tidx, avgp, acc);
  vq_gather  <<<T_TOK / 256, 256, 0, stream>>>(tidx, embn, zbn, out, acc);
  vq_finalize<<<1, 256, 0, stream>>>(avgp, acc, out);
}
